// WeibullDegeneracy_59871844106393
// MI455X (gfx1250) — matrix operand order repaired and checked by reading, not compiled, not run on hardware
//
#include <hip/hip_runtime.h>
#include <hip/hip_bf16.h>
#include <stdint.h>

typedef __attribute__((ext_vector_type(16))) _Float16 v16h;
typedef __attribute__((ext_vector_type(8)))  float    v8f;

typedef __attribute__((address_space(1))) int as1_int;
typedef __attribute__((address_space(3))) int as3_int;

#define NTHREADS 256
#define TILE     512   // 2 edges per thread, 16 (group,wave) chunks of 32 edges

#if defined(__has_builtin)
#if __has_builtin(__builtin_amdgcn_global_load_async_to_lds_b32)
#define HAVE_ASYNC_LDS 1
#endif
#endif

__device__ __forceinline__ float compute_ratio(const float* __restrict__ uw, int t) {
    // FREQUENCY = [0.5,0.3,0.7,0.2]; SCALE=1, CONCENTRATION=1 -> w = -log1p(-u)
    const float freq0 = 0.5f, freq1 = 0.3f, freq2 = 0.7f, freq3 = 0.2f;
    unsigned s = ((unsigned)t) & 3u;
    float f = (s == 0) ? freq0 : (s == 1) ? freq1 : (s == 2) ? freq2 : freq3;
    float w = -log1pf(-uw[0]);
    float r = fminf(fmaxf(f * w, 0.0f), 0.9f);   // clip to [0, 0.9]
    return (r < 0.1f) ? 0.0f : r;                // threshold
}

// ---------------- Pass 1: per-block active count + 32-edge ballot masks -------------
__global__ void wd_count(const float* __restrict__ aedges, const float* __restrict__ uedges,
                         const float* __restrict__ uweibull, const int* __restrict__ timep,
                         unsigned* __restrict__ blkCnt, unsigned* __restrict__ masks, int E)
{
    const int tid  = threadIdx.x;
    const int lane = tid & 31;
    const int tile = blockIdx.x * TILE;
    __shared__ unsigned bc;
    __shared__ float sA[TILE];
    __shared__ float sU[TILE];
    if (tid == 0) bc = 0;
    const float ratio = compute_ratio(uweibull, timep[0]);

#ifdef HAVE_ASYNC_LDS
    // CDNA5 async global -> LDS staging (ASYNCcnt-tracked, bypasses VGPRs)
    #pragma unroll
    for (int g = 0; g < 2; ++g) {
        int idx = g * NTHREADS + tid;
        int i   = tile + idx;
        if (i < E) {
            __builtin_amdgcn_global_load_async_to_lds_b32(
                (as1_int*)(aedges + i), (as3_int*)(&sA[idx]), 0, 0);
            __builtin_amdgcn_global_load_async_to_lds_b32(
                (as1_int*)(uedges + i), (as3_int*)(&sU[idx]), 0, 0);
        }
    }
#if defined(__has_builtin) && __has_builtin(__builtin_amdgcn_s_wait_asynccnt)
    __builtin_amdgcn_s_wait_asynccnt(0);
#else
    asm volatile("s_wait_asynccnt 0" ::: "memory");
#endif
#else
    #pragma unroll
    for (int g = 0; g < 2; ++g) {
        int idx = g * NTHREADS + tid;
        int i   = tile + idx;
        sA[idx] = (i < E) ? aedges[i] : 0.0f;
        sU[idx] = (i < E) ? uedges[i] : 1.0f;
    }
#endif
    __syncthreads();

    #pragma unroll
    for (int g = 0; g < 2; ++g) {
        int idx = g * NTHREADS + tid;
        int i   = tile + idx;
        // survives iff previously active AND not dropped (u >= ratio)
        bool active = (i < E) && (sA[idx] > 0.5f) && !(sU[idx] < ratio);
        unsigned mask = (unsigned)__ballot(active);   // wave32: low 32 bits
        if (lane == 0) {
            masks[(unsigned)i >> 5] = mask;
            atomicAdd(&bc, (unsigned)__popc(mask));
        }
    }
    __syncthreads();
    if (tid == 0) blkCnt[blockIdx.x] = bc;
}

// ---------------- Pass 2: exclusive scan of block counts (single block) -------------
__global__ void wd_scan(const unsigned* __restrict__ blkCnt, unsigned* __restrict__ blkScan, int nb)
{
    __shared__ unsigned wsum[8];
    __shared__ unsigned carry;
    const int tid = threadIdx.x, lane = tid & 31, w = tid >> 5;
    if (tid == 0) carry = 0;
    __syncthreads();
    for (int base = 0; base < nb; base += NTHREADS) {
        int i = base + tid;
        unsigned v = (i < nb) ? blkCnt[i] : 0u;
        unsigned x = v;
        #pragma unroll
        for (int d = 1; d < 32; d <<= 1) {
            unsigned y = __shfl_up(x, d, 32);
            if (lane >= d) x += y;
        }
        if (lane == 31) wsum[w] = x;
        __syncthreads();
        if (tid == 0) {
            unsigned run = carry;
            #pragma unroll
            for (int j = 0; j < 8; ++j) { unsigned t = wsum[j]; wsum[j] = run; run += t; }
            carry = run;
        }
        __syncthreads();
        if (i < nb) blkScan[i] = wsum[w] + (x - v);
        __syncthreads();
    }
    if (threadIdx.x == 0) blkScan[nb] = carry;   // total active
}

// ---------------- Pass 3: WMMA-scanned stable-partition scatter ---------------------
__global__ void wd_scatter(const float* __restrict__ edges,
                           const long long* __restrict__ rec, const long long* __restrict__ send,
                           const unsigned* __restrict__ masks, const unsigned* __restrict__ blkScan,
                           int nb, float* __restrict__ outNa, long long* __restrict__ outRec,
                           long long* __restrict__ outSend, float* __restrict__ outEdges,
                           long long sentinel, int E)
{
    const int tid  = threadIdx.x;
    const int lane = tid & 31;
    const int w    = tid >> 5;
    const int tile = blockIdx.x * TILE;

    __shared__ unsigned cnts[16];
    __shared__ unsigned sExcl[16];

    unsigned m[2];
    #pragma unroll
    for (int g = 0; g < 2; ++g) {
        int base = tile + g * NTHREADS + w * 32;        // 32-aligned, wave-uniform
        unsigned mask = masks[(unsigned)base >> 5];
        m[g] = mask;
        if (lane == 0) cnts[g * 8 + w] = (unsigned)__popc(mask);
    }
    __syncthreads();

    // Exclusive scan of the 16 chunk counts via one v_wmma_f32_16x16x32_f16:
    // D[m][n] = sum_k A[m][k]*B[k][n] with B[k][n]=(k<=n) -> inclusive scan along N.
    // Counts <= 32, partial sums <= 512: exact in f16. Wave 0 only, full EXEC.
    if (tid < 32) {
        v16h a, b;
        v8f  c = {};
        #pragma unroll
        for (int e = 0; e < 16; ++e) {
            // A 16x32 f16 layout: lane L<16 holds K 0..7 (halfs 0..7) / K16..23 (halfs 8..15);
            // lane L>=16 holds K 8..15 / K 24..31. Only K<16 carries data.
            int K = e + ((lane >= 16) ? 8 : 0);
            a[e] = (_Float16)((e < 8) ? (float)cnts[K] : 0.0f);
            // B 32x16 f16 (triangular ones): same K layout as A, N=lane%16; B[K][N]=(K<=N) for K<16.
            b[e] = (_Float16)((e < 8 && e + ((lane >= 16) ? 8 : 0) <= (lane & 15)) ? 1.0f : 0.0f);
        }
        v8f dacc = __builtin_amdgcn_wmma_f32_16x16x32_f16(
            false, a, false, b, (short)0, c, false, false);
        // D VGPR0, lanes 0..15 = row M=0, N=lane -> inclusive scan at position `lane`
        if (lane < 16) sExcl[lane] = (unsigned)(dacc[0] + 0.5f) - cnts[lane];
    }
    __syncthreads();

    const unsigned blockBase   = blkScan[blockIdx.x];
    const unsigned totalActive = blkScan[nb];

    #pragma unroll
    for (int g = 0; g < 2; ++g) {
        int i = tile + g * NTHREADS + tid;               // == chunkBase + lane
        if (i < E) {
            const float* srcRow = edges + (size_t)i * 16;
            __builtin_prefetch(srcRow, 0, 1);            // global_prefetch_b8
            unsigned mask   = m[g];
            bool     active = (mask >> lane) & 1u;
            unsigned p      = (unsigned)__popc(mask & ((1u << lane) - 1u));
            unsigned activeBefore = blockBase + sExcl[g * 8 + w] + p;
            unsigned dest = active ? activeBefore
                                   : totalActive + ((unsigned)i - activeBefore);
            outNa[dest]   = active ? 1.0f : 0.0f;
            outRec[dest]  = active ? rec[i]  : sentinel;
            outSend[dest] = active ? send[i] : sentinel;
            const float4* src = (const float4*)srcRow;
            float4*       dst = (float4*)(outEdges + (size_t)dest * 16);
            if (active) {
                float4 a0 = src[0], a1 = src[1], a2 = src[2], a3 = src[3];
                dst[0] = a0; dst[1] = a1; dst[2] = a2; dst[3] = a3;
            } else {
                float4 z = make_float4(0.0f, 0.0f, 0.0f, 0.0f);
                dst[0] = z; dst[1] = z; dst[2] = z; dst[3] = z;
            }
        }
    }
}

extern "C" void kernel_launch(void* const* d_in, const int* in_sizes, int n_in,
                              void* d_out, int out_size, void* d_ws, size_t ws_size,
                              hipStream_t stream) {
    // setup_inputs order: nodes, edges, aedges, u_weibull, u_edges, rec, send, time
    const float*     edges  = (const float*)d_in[1];
    const float*     aedges = (const float*)d_in[2];
    const float*     uweib  = (const float*)d_in[3];
    const float*     uedges = (const float*)d_in[4];
    const long long* rec    = (const long long*)d_in[5];
    const long long* send   = (const long long*)d_in[6];
    const int*       timep  = (const int*)d_in[7];

    const int E  = in_sizes[2];                 // number of edges
    const int N  = in_sizes[0] / 128;           // number of nodes (D_NODE = 128)
    const long long sentinel = (long long)N - 1;
    const int NB = (E + TILE - 1) / TILE;

    // Workspace: [blkCnt NB][blkScan NB+1][masks ceil(E/32)]
    unsigned* blkCnt  = (unsigned*)d_ws;
    unsigned* blkScan = blkCnt + NB;
    unsigned* masks   = blkScan + NB + 1;

    // Outputs byte-contiguous in return order:
    // naedges f32 [E] | nrec i64 [E] | nsend i64 [E] | new_edges f32 [E*16]
    float*     outNa    = (float*)d_out;
    long long* outRec   = (long long*)((char*)d_out + (size_t)4 * E);
    long long* outSend  = (long long*)((char*)d_out + (size_t)12 * E);
    float*     outEdges = (float*)((char*)d_out + (size_t)20 * E);

    wd_count<<<NB, NTHREADS, 0, stream>>>(aedges, uedges, uweib, timep, blkCnt, masks, E);
    wd_scan<<<1, NTHREADS, 0, stream>>>(blkCnt, blkScan, NB);
    wd_scatter<<<NB, NTHREADS, 0, stream>>>(edges, rec, send, masks, blkScan, NB,
                                            outNa, outRec, outSend, outEdges, sentinel, E);
}
